// UCCAEncoder_7301444403731
// MI455X (gfx1250) — compile-verified, matrix operand order checked by reading, not run on hardware
//
#include <hip/hip_runtime.h>

#define ND 8192
#define DD 512
#define EE 131072
#define BB 16
#define NPGc 512
#define SSc 128
#define LLc 3

#define WROW 40  // padded LDS row stride (elements) for a 32-wide weight row

typedef __attribute__((ext_vector_type(16))) __bf16 v16bf;
typedef __attribute__((ext_vector_type(8))) float v8f;
typedef __attribute__((ext_vector_type(4))) int v4i;

#if defined(__HIP_DEVICE_COMPILE__) &&                                    \
    __has_builtin(__builtin_amdgcn_global_load_async_to_lds_b128) &&      \
    __has_builtin(__builtin_amdgcn_s_wait_asynccnt)
#define USE_ASYNC 1
typedef __attribute__((address_space(1))) v4i* gv4i_p;
typedef __attribute__((address_space(3))) v4i* sv4i_p;
#else
#define USE_ASYNC 0
#endif

__device__ inline v8f wmma_bf16(v16bf a, v16bf b, v8f c) {
  return __builtin_amdgcn_wmma_f32_16x16x32_bf16(false, a, false, b, (short)0, c,
                                                 false, false);
}

// Load one lane's share of a 16x32 bf16 A/B fragment from a row-major row.
// Layout (ISA 05_wmma): lanes 0-15 -> K 0..7 & 16..23 ; lanes 16-31 -> +8.
__device__ inline v16bf ldfrag(const __bf16* p, int klo) {
  v16bf f;
  ((uint4*)&f)[0] = *(const uint4*)(p + klo);
  ((uint4*)&f)[1] = *(const uint4*)(p + klo + 16);
  return f;
}

// float atomic max via sign-split int atomics (init value must be -inf bits)
__device__ inline void atomicMaxF(float* a, float v) {
  if (v >= 0.f)
    atomicMax((int*)a, __float_as_int(v));
  else
    atomicMin((unsigned int*)a, __float_as_uint(v));
}

// Cooperative copy of one 256col x 32K bf16 weight tile (16 KB) into LDS.
// Packed weight layout: [K/32][512][32]; tile (kcb, g) is contiguous.
__device__ inline void stage_tile(__bf16* __restrict__ dst,
                                  const __bf16* __restrict__ wpk, int kcb,
                                  int g) {
  const int c = threadIdx.x;  // 0..255 -> one output column each
  const __bf16* src = wpk + ((size_t)kcb * DD + g * 256 + c) * 32;
  __bf16* d = dst + (size_t)c * WROW;
#if USE_ASYNC
#pragma unroll
  for (int i = 0; i < 4; ++i)
    __builtin_amdgcn_global_load_async_to_lds_b128(
        (gv4i_p)(src + i * 8), (sv4i_p)(d + i * 8), 0, 0);
#else
  const uint4* s4 = (const uint4*)src;
  uint4 t0 = s4[0], t1 = s4[1], t2 = s4[2], t3 = s4[3];
  uint4* d4 = (uint4*)d;
  d4[0] = t0;
  d4[1] = t1;
  d4[2] = t2;
  d4[3] = t3;
#endif
}

__device__ inline void wait_staging() {
#if USE_ASYNC
  __builtin_amdgcn_s_wait_asynccnt(0);
#endif
}

// 16 col-tiles: batched ds_read_b128 clauses then back-to-back WMMAs
__device__ inline void wmma16(const __bf16* __restrict__ wb, int r, int klo,
                              v16bf a, v8f acc[16]) {
#pragma unroll
  for (int tb = 0; tb < 2; ++tb) {
    v16bf bfr[8];
#pragma unroll
    for (int u = 0; u < 8; ++u)
      bfr[u] = ldfrag(wb + (size_t)((tb * 8 + u) * 16 + r) * WROW, klo);
#pragma unroll
    for (int u = 0; u < 8; ++u)
      acc[tb * 8 + u] = wmma_bf16(a, bfr[u], acc[tb * 8 + u]);
  }
}

// ---------------- LayerNorm -> bf16 node table ----------------
__global__ __launch_bounds__(256) void k_ln(const float* __restrict__ x,
                                            const float* __restrict__ g,
                                            const float* __restrict__ b,
                                            __bf16* __restrict__ h) {
  const int lane = threadIdx.x & 31;
  const int wv = threadIdx.x >> 5;
  const int row = blockIdx.x * 8 + wv;
  const float* xr = x + (size_t)row * DD;
  float v[16];
  float s = 0.f, ss = 0.f;
#pragma unroll
  for (int i = 0; i < 16; ++i) {
    v[i] = xr[lane + 32 * i];
    s += v[i];
    ss += v[i] * v[i];
  }
#pragma unroll
  for (int off = 16; off >= 1; off >>= 1) {
    s += __shfl_xor(s, off, 32);
    ss += __shfl_xor(ss, off, 32);
  }
  const float mu = s * (1.f / (float)DD);
  const float var = ss * (1.f / (float)DD) - mu * mu;
  const float rs = rsqrtf(var + 1e-5f);
  __bf16* hr = h + (size_t)row * DD;
#pragma unroll
  for (int i = 0; i < 16; ++i) {
    const int c = lane + 32 * i;
    hr[c] = (__bf16)((v[i] - mu) * rs * g[c] + b[c]);
  }
}

__global__ void k_fill_neginf(unsigned int* p, int n) {
  int i = blockIdx.x * blockDim.x + threadIdx.x;
  if (i < n) p[i] = 0xFF800000u;
}

__global__ void k_finalize(const float* __restrict__ agg,
                           const float* __restrict__ resid,
                           float* __restrict__ xout, int n) {
  int i = blockIdx.x * blockDim.x + threadIdx.x;
  if (i < n) xout[i] = fmaxf(agg[i], 0.f) + resid[i];
}

// weight prep: fp32 (K,Nn) row-major -> bf16 packed [K/32][Nn][32]
__global__ void k_wprep(const float* __restrict__ src, __bf16* __restrict__ dst,
                        int K, int Nn) {
  int i = blockIdx.x * blockDim.x + threadIdx.x;
  if (i >= K * Nn) return;
  int n = i % Nn, k = i / Nn;
  dst[((size_t)(k >> 5) * Nn + n) * 32 + (k & 31)] = (__bf16)src[i];
}

// ---------------- EdgeConv: fused 2-stage WMMA GEMM + atomic segment-max ----
__global__ __launch_bounds__(256) void k_edge(
    const __bf16* __restrict__ h, const int* __restrict__ esrc,
    const int* __restrict__ edst, const __bf16* __restrict__ w1p,
    const float* __restrict__ b1, const __bf16* __restrict__ w2p,
    const float* __restrict__ b2, float* __restrict__ agg) {
  extern __shared__ char smem[];
  __bf16* hid0 = (__bf16*)smem;                         // 8 * 16 * 512 bf16
  __bf16* wbuf0 = (__bf16*)(smem + 8 * 16 * DD * 2);    // 256 * WROW bf16
  __bf16* wbuf1 = wbuf0 + 256 * WROW;

  const int lane = threadIdx.x & 31;
  const int wv = threadIdx.x >> 5;
  __bf16* hid = hid0 + (size_t)wv * 16 * DD;
  const int r = lane & 15;
  const int half = lane >> 4;
  const int klo = half * 8;
  const int ebase = (blockIdx.x * 8 + wv) * 16;
  const int dn = edst[ebase + r];
  const int sn = esrc[ebase + r];
  const __bf16* hi = h + (size_t)dn * DD;
  const __bf16* hj = h + (size_t)sn * DD;

  // stage 1: hidden = relu([h_i , h_j - h_i] @ W1 + b1) -> LDS bf16 (16x512)
  for (int g = 0; g < 2; ++g) {
    v8f acc[16];
    v8f z = {};
#pragma unroll
    for (int t = 0; t < 16; ++t) acc[t] = z;
    stage_tile(wbuf0, w1p, 0, g);
    for (int kcb = 0; kcb < 32; ++kcb) {
      __bf16* cw = (kcb & 1) ? wbuf1 : wbuf0;
      wait_staging();
      __syncthreads();
      if (kcb + 1 < 32) stage_tile((kcb & 1) ? wbuf0 : wbuf1, w1p, kcb + 1, g);
      const int kc = kcb * 32;
      v16bf a;
      if (kc < DD) {
        a = ldfrag(hi + kc, klo);
      } else {
        v16bf aj = ldfrag(hj + (kc - DD), klo);
        v16bf ai = ldfrag(hi + (kc - DD), klo);
#pragma unroll
        for (int q = 0; q < 16; ++q)
          a[q] = (__bf16)((float)aj[q] - (float)ai[q]);
      }
      wmma16(cw, r, klo, a, acc);
    }
#pragma unroll
    for (int t = 0; t < 16; ++t) {
      const int col = (g * 16 + t) * 16 + r;
      const float bias = b1[col];
#pragma unroll
      for (int reg = 0; reg < 8; ++reg)
        hid[(size_t)(reg + 8 * half) * DD + col] =
            (__bf16)fmaxf(acc[t][reg] + bias, 0.f);
    }
  }
  __syncthreads();

  // stage 2: m = hidden @ W2 + b2 ; atomic-max scatter into agg[dst]
  for (int g = 0; g < 2; ++g) {
    v8f acc[16];
    v8f z = {};
#pragma unroll
    for (int t = 0; t < 16; ++t) acc[t] = z;
    stage_tile(wbuf0, w2p, 0, g);
    for (int kcb = 0; kcb < 16; ++kcb) {
      __bf16* cw = (kcb & 1) ? wbuf1 : wbuf0;
      wait_staging();
      __syncthreads();
      if (kcb + 1 < 16) stage_tile((kcb & 1) ? wbuf0 : wbuf1, w2p, kcb + 1, g);
      v16bf a = ldfrag(hid + (size_t)r * DD + kcb * 32, klo);
      wmma16(cw, r, klo, a, acc);
    }
#pragma unroll
    for (int t = 0; t < 16; ++t) {
      const int col = (g * 16 + t) * 16 + r;
      const float bias = b2[col];
#pragma unroll
      for (int reg = 0; reg < 8; ++reg) {
        const int m = reg + 8 * half;
        const int drow = __shfl(dn, m, 32);
        atomicMaxF(agg + (size_t)drow * DD + col, acc[t][reg] + bias);
      }
    }
  }
}

// ---------------- final FFN: x += relu(h @ Wf1 + b1) @ Wf2 + b2 --------------
__global__ __launch_bounds__(256) void k_ffn(
    const __bf16* __restrict__ h, const __bf16* __restrict__ w1p,
    const float* __restrict__ b1, const __bf16* __restrict__ w2p,
    const float* __restrict__ b2, float* __restrict__ x) {
  extern __shared__ char smem[];
  __bf16* hid0 = (__bf16*)smem;
  __bf16* wbuf0 = (__bf16*)(smem + 8 * 16 * DD * 2);
  __bf16* wbuf1 = wbuf0 + 256 * WROW;

  const int lane = threadIdx.x & 31;
  const int wv = threadIdx.x >> 5;
  __bf16* hid = hid0 + (size_t)wv * 16 * DD;
  const int r = lane & 15;
  const int half = lane >> 4;
  const int klo = half * 8;
  const int nbase = (blockIdx.x * 8 + wv) * 16;
  const __bf16* hr = h + (size_t)(nbase + r) * DD;

  for (int g = 0; g < 2; ++g) {
    v8f acc[16];
    v8f z = {};
#pragma unroll
    for (int t = 0; t < 16; ++t) acc[t] = z;
    stage_tile(wbuf0, w1p, 0, g);
    for (int kcb = 0; kcb < 16; ++kcb) {
      __bf16* cw = (kcb & 1) ? wbuf1 : wbuf0;
      wait_staging();
      __syncthreads();
      if (kcb + 1 < 16) stage_tile((kcb & 1) ? wbuf0 : wbuf1, w1p, kcb + 1, g);
      v16bf a = ldfrag(hr + kcb * 32, klo);
      wmma16(cw, r, klo, a, acc);
    }
#pragma unroll
    for (int t = 0; t < 16; ++t) {
      const int col = (g * 16 + t) * 16 + r;
      const float bias = b1[col];
#pragma unroll
      for (int reg = 0; reg < 8; ++reg)
        hid[(size_t)(reg + 8 * half) * DD + col] =
            (__bf16)fmaxf(acc[t][reg] + bias, 0.f);
    }
  }
  __syncthreads();

  for (int g = 0; g < 2; ++g) {
    v8f acc[16];
    v8f z = {};
#pragma unroll
    for (int t = 0; t < 16; ++t) acc[t] = z;
    stage_tile(wbuf0, w2p, 0, g);
    for (int kcb = 0; kcb < 16; ++kcb) {
      __bf16* cw = (kcb & 1) ? wbuf1 : wbuf0;
      wait_staging();
      __syncthreads();
      if (kcb + 1 < 16) stage_tile((kcb & 1) ? wbuf0 : wbuf1, w2p, kcb + 1, g);
      v16bf a = ldfrag(hid + (size_t)r * DD + kcb * 32, klo);
      wmma16(cw, r, klo, a, acc);
    }
#pragma unroll
    for (int t = 0; t < 16; ++t) {
      const int col = (g * 16 + t) * 16 + r;
      const float bias = b2[col];
#pragma unroll
      for (int reg = 0; reg < 8; ++reg) {
        const int row = nbase + reg + 8 * half;
        const size_t off = (size_t)row * DD + col;
        x[off] = acc[t][reg] + bias + x[off];
      }
    }
  }
}

// ---------------- gather selected nodes -> out [B,S,D] fp32 ------------------
__global__ void k_gather(const float* __restrict__ x, const int* __restrict__ sel,
                         float* __restrict__ out) {
  const int rowo = blockIdx.x;  // 0..B*S-1
  const int bgr = rowo / SSc;
  const int node = bgr * NPGc + sel[rowo];
  const float* sp = x + (size_t)node * DD;
  float* dp = out + (size_t)rowo * DD;
  for (int c = threadIdx.x; c < DD; c += blockDim.x) dp[c] = sp[c];
}

extern "C" void kernel_launch(void* const* d_in, const int* in_sizes, int n_in,
                              void* d_out, int out_size, void* d_ws,
                              size_t ws_size, hipStream_t stream) {
  (void)in_sizes; (void)n_in; (void)out_size; (void)ws_size;
  const float* x_in = (const float*)d_in[0];
  const int* eidx = (const int*)d_in[1];
  const int* sel = (const int*)d_in[2];
  const float* cW1 = (const float*)d_in[4];
  const float* cb1 = (const float*)d_in[5];
  const float* cW2 = (const float*)d_in[6];
  const float* cb2 = (const float*)d_in[7];
  const float* ln1g = (const float*)d_in[8];
  const float* ln1b = (const float*)d_in[9];
  const float* ln2g = (const float*)d_in[10];
  const float* ln2b = (const float*)d_in[11];
  const float* fW1 = (const float*)d_in[12];
  const float* fb1 = (const float*)d_in[13];
  const float* fW2 = (const float*)d_in[14];
  const float* fb2 = (const float*)d_in[15];
  float* outp = (float*)d_out;

  const int* esrc = eidx;       // edge_index[0] = x_j
  const int* edst = eidx + EE;  // edge_index[1] = x_i / aggregation index

  char* w = (char*)d_ws;
  __bf16* hbf = (__bf16*)w;  w += (size_t)ND * DD * 2;
  float* xbuf = (float*)w;   w += (size_t)ND * DD * 4;
  float* agg = (float*)w;    w += (size_t)ND * DD * 4;
  __bf16* wt1 = (__bf16*)w;  w += (size_t)LLc * 2 * DD * DD * 2;
  __bf16* wt2 = (__bf16*)w;  w += (size_t)LLc * DD * DD * 2;
  __bf16* wf1 = (__bf16*)w;  w += (size_t)DD * DD * 2;
  __bf16* wf2 = (__bf16*)w;  w += (size_t)DD * DD * 2;

  // weight prep: bf16 + pack into [K/32][N][32] so LDS tiles are contiguous
  for (int l = 0; l < LLc; ++l) {
    k_wprep<<<(2 * DD * DD + 255) / 256, 256, 0, stream>>>(
        cW1 + (size_t)l * 2 * DD * DD, wt1 + (size_t)l * DD * 2 * DD, 2 * DD, DD);
    k_wprep<<<(DD * DD + 255) / 256, 256, 0, stream>>>(
        cW2 + (size_t)l * DD * DD, wt2 + (size_t)l * DD * DD, DD, DD);
  }
  k_wprep<<<(DD * DD + 255) / 256, 256, 0, stream>>>(fW1, wf1, DD, DD);
  k_wprep<<<(DD * DD + 255) / 256, 256, 0, stream>>>(fW2, wf2, DD, DD);

  // LDS: 128 KB hidden + 2 x 20 KB double-buffered weight tiles
  const size_t ldsz = (size_t)8 * 16 * DD * 2 + 2 * 256 * WROW * 2;

  k_ln<<<ND / 8, 256, 0, stream>>>(x_in, ln1g, ln1b, hbf);
  for (int l = 0; l < LLc; ++l) {
    k_fill_neginf<<<(ND * DD + 255) / 256, 256, 0, stream>>>(
        (unsigned int*)agg, ND * DD);
    k_edge<<<EE / (16 * 8), 256, ldsz, stream>>>(
        hbf, esrc, edst, wt1 + (size_t)l * DD * 2 * DD, cb1 + (size_t)l * DD,
        wt2 + (size_t)l * DD * DD, cb2 + (size_t)l * DD, agg);
    const float* resid = (l == 0) ? x_in : xbuf;
    k_finalize<<<(ND * DD + 255) / 256, 256, 0, stream>>>(agg, resid, xbuf,
                                                          ND * DD);
    if (l < LLc - 1)
      k_ln<<<ND / 8, 256, 0, stream>>>(xbuf, ln1g, ln1b, hbf);
    else
      k_ln<<<ND / 8, 256, 0, stream>>>(xbuf, ln2g, ln2b, hbf);
  }
  k_ffn<<<ND / (16 * 8), 256, ldsz, stream>>>(hbf, wf1, fb1, wf2, fb2, xbuf);
  k_gather<<<BB * SSc, 256, 0, stream>>>(xbuf, sel, outp);
}
